// DeformDDPM_10256381903142
// MI455X (gfx1250) — matrix operand order repaired and checked by reading, not compiled, not run on hardware
//
#include <hip/hip_runtime.h>
#include <math.h>

typedef __attribute__((ext_vector_type(16))) _Float16 v16h;
typedef __attribute__((ext_vector_type(8)))  float    v8f;

#define CTL    64
#define CTL2   4096
#define IMG    256
#define IMGPIX 65536
#define NTHR   1024

__device__ __forceinline__ int   imin(int a, int b)   { return a < b ? a : b; }
__device__ __forceinline__ int   imax(int a, int b)   { return a > b ? a : b; }
__device__ __forceinline__ float clampf(float v, float lo, float hi) {
  return fminf(fmaxf(v, lo), hi);
}

struct Bil4 { int i00, i01, i10, i11; float w00, w01, w10, w11; };

// border-mode bilinear taps on a 64x64 plane
__device__ __forceinline__ Bil4 make_bil_border64(float cy, float cx) {
  cy = clampf(cy, 0.0f, 63.0f);
  cx = clampf(cx, 0.0f, 63.0f);
  float yf = floorf(cy), xf = floorf(cx);
  int y0 = (int)yf, x0 = (int)xf;
  float wy = cy - yf, wx = cx - xf;
  int y1 = imin(y0 + 1, 63), x1 = imin(x0 + 1, 63);
  Bil4 b;
  b.i00 = y0 * CTL + x0; b.i01 = y0 * CTL + x1;
  b.i10 = y1 * CTL + x0; b.i11 = y1 * CTL + x1;
  b.w00 = (1.0f - wy) * (1.0f - wx); b.w01 = (1.0f - wy) * wx;
  b.w10 = wy * (1.0f - wx);          b.w11 = wy * wx;
  return b;
}

__device__ __forceinline__ float bil_apply(const float* p, const Bil4& b) {
  return p[b.i00] * b.w00 + p[b.i01] * b.w01 + p[b.i10] * b.w10 + p[b.i11] * b.w11;
}

// half-pixel (align_corners=False) bilinear sample of an s x s plane at
// output pixel (y,x) of the 64x64 grid, with edge clamping (== jax resize).
__device__ __forceinline__ float up_sample(const float* __restrict__ src, int s, int y, int x) {
  float sc = (float)s * (1.0f / 64.0f);
  float sy = ((float)y + 0.5f) * sc - 0.5f;
  float sx = ((float)x + 0.5f) * sc - 0.5f;
  float mx = (float)(s - 1);
  sy = clampf(sy, 0.0f, mx);
  sx = clampf(sx, 0.0f, mx);
  float yf = floorf(sy), xf = floorf(sx);
  int y0 = (int)yf, x0 = (int)xf;
  float fy = sy - yf, fx = sx - xf;
  int y1 = imin(y0 + 1, s - 1), x1 = imin(x0 + 1, s - 1);
  float v00 = src[y0 * s + x0], v01 = src[y0 * s + x1];
  float v10 = src[y1 * s + x0], v11 = src[y1 * s + x1];
  return v00 * (1.0f - fy) * (1.0f - fx) + v01 * (1.0f - fy) * fx
       + v10 * fy * (1.0f - fx) + v11 * fy * fx;
}

__device__ __forceinline__ float multiscale(const float* __restrict__ a64,
                                            const float* __restrict__ a32,
                                            const float* __restrict__ a16,
                                            const float* __restrict__ a8,
                                            int y, int x, float vs) {
  float r = up_sample(a64, 64, y, x) * vs;          // identity at s=64
  r += up_sample(a32, 32, y, x) * (vs * 2.0f);
  r += up_sample(a16, 16, y, x) * (vs * 4.0f);
  r += up_sample(a8,   8, y, x) * (vs * 8.0f);
  return r;
}

// zeros-padding tap on a 256x256 image
__device__ __forceinline__ float tapz(const float* __restrict__ img, int iy, int ix) {
  int yc = imin(imax(iy, 0), 255), xc = imin(imax(ix, 0), 255);
  float v = img[yc * IMG + xc];
  bool ok = (iy >= 0) && (iy <= 255) && (ix >= 0) && (ix <= 255);
  return ok ? v : 0.0f;
}

extern "C" __global__ __launch_bounds__(NTHR, 1)
void deform_ddpm_kernel(const float* __restrict__ x0,
                        const int*   __restrict__ tptr,
                        const float* __restrict__ n64, const float* __restrict__ n32,
                        const float* __restrict__ n16, const float* __restrict__ n8,
                        const float* __restrict__ e64, const float* __restrict__ e32,
                        const float* __restrict__ e16, const float* __restrict__ e8,
                        float* __restrict__ out) {
  extern __shared__ float sm[];
  float*    nse  = sm;                 // [2*4096] fp32   (phase A/B only)
  float*    ddf  = sm + 2 * CTL2;      // [2*4096] fp32
  float*    dddf = sm + 4 * CTL2;      // [2*4096] fp32
  float*    Tbuf = sm + 6 * CTL2;      // [256*64] fp32   (GEMM intermediate)
  _Float16* Uw   = (_Float16*)sm;      // [256*64] f16 upsample weights, aliases nse

  const int tid = threadIdx.x;
  const int b   = blockIdx.x;

  // ---- iteration counts from t (uniform across block) -----------------
  const double td   = (double)tptr[0];
  const int n_ddf   = (int)fmin(fmax(floor(2.0 * pow(td, 1.3) / 3.0), 1.0), 200.0);
  const int n_dvf   = (int)floor(pow(td, 0.6));

  const float V_SCALE = 0.008f / 256.0f;
  const float NR      = 0.08f;
  const int   b2      = b * 2;

  // ---- Phase A: multiscale dvf (regs) + nse (LDS) ---------------------
  float dvfy[4], dvfx[4], dvf0y[4], dvf0x[4];
#pragma unroll
  for (int k = 0; k < 4; k++) {
    int p = tid + k * NTHR, y = p >> 6, x = p & 63;
    dvfy[k] = multiscale(n64 + (size_t)(b2 + 0) * 4096, n32 + (size_t)(b2 + 0) * 1024,
                         n16 + (size_t)(b2 + 0) * 256,  n8  + (size_t)(b2 + 0) * 64,
                         y, x, V_SCALE);
    dvfx[k] = multiscale(n64 + (size_t)(b2 + 1) * 4096, n32 + (size_t)(b2 + 1) * 1024,
                         n16 + (size_t)(b2 + 1) * 256,  n8  + (size_t)(b2 + 1) * 64,
                         y, x, V_SCALE);
    nse[p]        = multiscale(e64 + (size_t)(b2 + 0) * 4096, e32 + (size_t)(b2 + 0) * 1024,
                               e16 + (size_t)(b2 + 0) * 256,  e8  + (size_t)(b2 + 0) * 64,
                               y, x, V_SCALE * NR);
    nse[CTL2 + p] = multiscale(e64 + (size_t)(b2 + 1) * 4096, e32 + (size_t)(b2 + 1) * 1024,
                               e16 + (size_t)(b2 + 1) * 256,  e8  + (size_t)(b2 + 1) * 64,
                               y, x, V_SCALE * NR);
  }
  __syncthreads();

  // ---- Phase B: dvf0 = dvf + warp(nse, dvf, border); init ddf/dddf ----
#pragma unroll
  for (int k = 0; k < 4; k++) {
    int p = tid + k * NTHR, y = p >> 6, x = p & 63;
    Bil4 bb = make_bil_border64((float)y + dvfy[k], (float)x + dvfx[k]);
    dvf0y[k] = dvfy[k] + bil_apply(nse, bb);
    dvf0x[k] = dvfx[k] + bil_apply(nse + CTL2, bb);
    ddf[p] = 0.0f;  ddf[CTL2 + p] = 0.0f;
    dddf[p] = 0.0f; dddf[CTL2 + p] = 0.0f;
  }
  __syncthreads();

  // ---- Phase C: sequential recurrence, fields live in LDS -------------
  // dddf update is exact identity once j >= n_dvf (zero displacement), skip it.
  float sy_[4], sx_[4], ty_[4], tx_[4];
  for (int j = 0; j < n_ddf; j++) {
    bool upd = (j < n_dvf);
#pragma unroll
    for (int k = 0; k < 4; k++) {
      int p = tid + k * NTHR, y = p >> 6, x = p & 63;
      Bil4 bb = make_bil_border64((float)y + dvf0y[k], (float)x + dvf0x[k]);
      sy_[k] = dvf0y[k] + bil_apply(ddf, bb);
      sx_[k] = dvf0x[k] + bil_apply(ddf + CTL2, bb);
    }
    if (upd) {
#pragma unroll
      for (int k = 0; k < 4; k++) {
        int p = tid + k * NTHR, y = p >> 6, x = p & 63;
        Bil4 bb = make_bil_border64((float)y + dvfy[k], (float)x + dvfx[k]);
        ty_[k] = dvfy[k] + bil_apply(dddf, bb);
        tx_[k] = dvfx[k] + bil_apply(dddf + CTL2, bb);
      }
    }
    __syncthreads();
#pragma unroll
    for (int k = 0; k < 4; k++) {
      int p = tid + k * NTHR;
      ddf[p] = sy_[k]; ddf[CTL2 + p] = sx_[k];
    }
    if (upd) {
#pragma unroll
      for (int k = 0; k < 4; k++) {
        int p = tid + k * NTHR;
        dddf[p] = ty_[k]; dddf[CTL2 + p] = tx_[k];
      }
    }
    __syncthreads();
  }

  // ---- Phase D: upsample_crop via WMMA:  O = 4 * (U . F . U^T) --------
  // U[oy,y]: 2-tap bilinear weights; src coord sy=(oy+128.5)/8-0.5, all
  // interior (15.56..47.44). Weights are k/16 => exact in f16.
  for (int i = tid; i < IMG * CTL; i += NTHR) {
    int oy = i >> 6, yy = i & 63;
    float sy = ((float)oy + 124.5f) * 0.125f;
    int y0 = (int)sy;
    float fy = sy - (float)y0;
    float w = (yy == y0) ? (1.0f - fy) : ((yy == y0 + 1) ? fy : 0.0f);
    Uw[i] = (_Float16)w;                     // aliases nse region (dead now)
  }
  __syncthreads();

  const int lane = tid & 31;
  const int wv   = tid >> 5;       // 32 waves
  const int Mh   = lane & 15;
  const int hb   = lane >> 4;      // half-wave selector

  for (int fc = 0; fc < 4; fc++) {
    const float* F = ((fc < 2) ? dddf : ddf) + (fc & 1) * CTL2;
    // out layout: warped @0, dvf_full(=up(dddf)) @4194304, ddf_full(=up(ddf)) @12582912
    size_t outBase = ((fc < 2) ? (size_t)4194304 : (size_t)12582912)
                   + (size_t)(b * 2 + (fc & 1)) * (size_t)IMGPIX;

    // GEMM1: T[256x64] = U[256x64] * F[64x64], F split into f16 hi+lo
    for (int tile = wv; tile < 64; tile += 32) {
      int tm = tile >> 2, tn = tile & 3;
      v8f acc = {};
#pragma unroll
      for (int ks = 0; ks < 2; ks++) {
        int k0 = ks * 32;
        v16h a, bhv, blv;
#pragma unroll
        for (int i = 0; i < 16; i++) {   // A frag: M=lane&15, K per ISA layout
          int K = (hb << 3) + ((i >> 3) << 4) + (i & 7);
          a[i] = Uw[(tm * 16 + Mh) * 64 + k0 + K];
        }
#pragma unroll
        for (int i = 0; i < 16; i++) {   // B frag: N=lane&15, K per ISA layout
          int K = (hb << 3) + ((i >> 3) << 4) + (i & 7);
          float v = F[(k0 + K) * 64 + tn * 16 + Mh];
          _Float16 h = (_Float16)v;
          bhv[i] = h;
          blv[i] = (_Float16)(v - (float)h);
        }
        acc = __builtin_amdgcn_wmma_f32_16x16x32_f16(false, a, false, bhv, (short)0, acc, false, false);
        acc = __builtin_amdgcn_wmma_f32_16x16x32_f16(false, a, false, blv, (short)0, acc, false, false);
      }
#pragma unroll
      for (int r = 0; r < 8; r++)
        Tbuf[(tm * 16 + hb * 8 + r) * 64 + tn * 16 + Mh] = acc[r];
    }
    __syncthreads();

    // GEMM2: O[256x256] = T[256x64] * U^T[64x256], T split hi+lo; store 4*O
    for (int tile = wv; tile < 256; tile += 32) {
      int tm = tile >> 4, tn = tile & 15;
      v8f acc = {};
#pragma unroll
      for (int ks = 0; ks < 2; ks++) {
        int k0 = ks * 32;
        v16h ah, al, bbv;
#pragma unroll
        for (int i = 0; i < 16; i++) {
          int K = (hb << 3) + ((i >> 3) << 4) + (i & 7);
          float v = Tbuf[(tm * 16 + Mh) * 64 + k0 + K];
          _Float16 h = (_Float16)v;
          ah[i] = h;
          al[i] = (_Float16)(v - (float)h);
        }
#pragma unroll
        for (int i = 0; i < 16; i++) {   // B = U^T: element (K,N) = U[N,K]
          int K = (hb << 3) + ((i >> 3) << 4) + (i & 7);
          bbv[i] = Uw[(tn * 16 + Mh) * 64 + k0 + K];
        }
        acc = __builtin_amdgcn_wmma_f32_16x16x32_f16(false, ah, false, bbv, (short)0, acc, false, false);
        acc = __builtin_amdgcn_wmma_f32_16x16x32_f16(false, al, false, bbv, (short)0, acc, false, false);
      }
#pragma unroll
      for (int r = 0; r < 8; r++)
        out[outBase + (size_t)(tm * 16 + hb * 8 + r) * IMG + tn * 16 + Mh] = 4.0f * acc[r];
    }
    __syncthreads();
  }

  // ---- Phase E: warped = warp(x_0, ddf_full, 'zeros') -----------------
  // ddf_full needed only per-pixel: exact 4-tap bilerp of LDS ddf, x4.
  const float* x0b = x0 + (size_t)b * IMGPIX;
  for (int p = tid; p < IMGPIX; p += NTHR) {
    int oy = p >> 8, ox = p & 255;
    __builtin_prefetch(x0b + p, 0, 1);
    float syf = ((float)oy + 124.5f) * 0.125f;
    float sxf = ((float)ox + 124.5f) * 0.125f;
    int y0 = (int)syf, xq = (int)sxf;            // interior: 15..47
    float fy = syf - (float)y0, fx = sxf - (float)xq;
    int i00 = y0 * CTL + xq, i01 = i00 + 1, i10 = i00 + CTL, i11 = i10 + 1;
    float w00 = (1.0f - fy) * (1.0f - fx), w01 = (1.0f - fy) * fx;
    float w10 = fy * (1.0f - fx),          w11 = fy * fx;
    float dy = 4.0f * (ddf[i00] * w00 + ddf[i01] * w01 + ddf[i10] * w10 + ddf[i11] * w11);
    float dx = 4.0f * (ddf[CTL2 + i00] * w00 + ddf[CTL2 + i01] * w01
                     + ddf[CTL2 + i10] * w10 + ddf[CTL2 + i11] * w11);
    float cy = (float)oy + dy, cx = (float)ox + dx;   // no clip: zeros padding
    float yfl = floorf(cy), xfl = floorf(cx);
    int iy = (int)yfl, ix = (int)xfl;
    float wy = cy - yfl, wx = cx - xfl;
    float v00 = tapz(x0b, iy, ix),     v01 = tapz(x0b, iy, ix + 1);
    float v10 = tapz(x0b, iy + 1, ix), v11 = tapz(x0b, iy + 1, ix + 1);
    out[(size_t)b * IMGPIX + p] =
        v00 * (1.0f - wy) * (1.0f - wx) + v01 * (1.0f - wy) * wx
      + v10 * wy * (1.0f - wx) + v11 * wy * wx;
  }
}

extern "C" void kernel_launch(void* const* d_in, const int* in_sizes, int n_in,
                              void* d_out, int out_size, void* d_ws, size_t ws_size,
                              hipStream_t stream) {
  const float* x0  = (const float*)d_in[0];
  const int*   tp  = (const int*)d_in[1];
  const float* n64 = (const float*)d_in[2];
  const float* n32 = (const float*)d_in[3];
  const float* n16 = (const float*)d_in[4];
  const float* n8  = (const float*)d_in[5];
  const float* e64 = (const float*)d_in[6];
  const float* e32 = (const float*)d_in[7];
  const float* e16 = (const float*)d_in[8];
  const float* e8  = (const float*)d_in[9];
  float* out = (float*)d_out;

  const int B = in_sizes[0] / (IMG * IMG);   // 64
  const size_t shmem = 10 * CTL2 * sizeof(float) + IMG * CTL * sizeof(float); // 224 KB
  (void)hipFuncSetAttribute((const void*)deform_ddpm_kernel,
                            hipFuncAttributeMaxDynamicSharedMemorySize, (int)shmem);
  deform_ddpm_kernel<<<dim3(B), dim3(NTHR), shmem, stream>>>(
      x0, tp, n64, n32, n16, n8, e64, e32, e16, e8, out);
}
